// FPERouter_84181359001986
// MI455X (gfx1250) — matrix operand order repaired and checked by reading, not compiled, not run on hardware
//
#include <hip/hip_runtime.h>
#include <hip/hip_bf16.h>
#include <math.h>
#include <stdint.h>

// -------------------------------------------------------------------------
// FPE router for MI455X (gfx1250).
//   scores = l2norm( (X@W^T) @ C ) @ E^T   with C circulant from R.
// l2norm of x_proj cancels (scores invariant to row scale), so
//   scores[n,:] = (Y @ (C E^T))[n,:] / ||Y[n] @ C||,   Y = X@W^T.
// Big compute = two 16384x4096x4096 GEMMs -> v_wmma_f32_16x16x32_bf16.
// Staging: A tile via Tensor Data Mover (tensor_load_to_lds, TENSORcnt,
// LDS pad feature gives the stride-40 layout), B tile via per-lane async
// global->LDS DMA (ASYNCcnt). Both double buffered; last K-step peeled so
// the steady-state loop body is branch-free.
// -------------------------------------------------------------------------

#define N_TOK 16384
#define DIMV  4096
#define HID   4096
#define NEXP  64

typedef __bf16 bf16_t;
typedef bf16_t v16bf __attribute__((ext_vector_type(16)));
typedef bf16_t v8bf  __attribute__((ext_vector_type(8)));
typedef float  v8f   __attribute__((ext_vector_type(8)));
typedef unsigned v4u __attribute__((ext_vector_type(4)));
typedef unsigned v8u __attribute__((ext_vector_type(8)));

// build fragment from two 16B LDS halves (two ds_load_b128)
__device__ __forceinline__ v16bf load_frag16_2(const bf16_t* plo, const bf16_t* phi) {
  v8bf lo = *(const v8bf*)plo;
  v8bf hi = *(const v8bf*)phi;
  v16bf r;
#pragma unroll
  for (int e = 0; e < 8; ++e) { r[e] = lo[e]; r[e + 8] = hi[e]; }
  return r;
}

// async 16B global->LDS copy (no VGPR round-trip); tracked by ASYNCcnt
__device__ __forceinline__ void async_copy16(unsigned lds_byte_addr,
                                             const void* gptr, int ioff) {
  const unsigned long long ga = (unsigned long long)(uintptr_t)gptr;
  if (ioff == 0)
    asm volatile("global_load_async_to_lds_b128 %0, %1, off"
                 :: "v"(lds_byte_addr), "v"(ga) : "memory");
  else
    asm volatile("global_load_async_to_lds_b128 %0, %1, off offset:16"
                 :: "v"(lds_byte_addr), "v"(ga) : "memory");
}

// TDM: DMA a 2D bf16 tile (tile_x elems per row, tile_y rows, row stride
// in elements) from global to LDS, padding LDS by 16B after each 64B row
// (=> LDS row stride 40 bf16). One instruction moves the whole tile.
__device__ __forceinline__ void tdm_load_2d(unsigned lds_addr, const void* gptr,
                                            unsigned tile_x, unsigned tile_y,
                                            unsigned row_stride_elems) {
  const unsigned long long ga = (unsigned long long)(uintptr_t)gptr;
  v4u g0;
  g0[0] = 1u;                                          // count=1, user D#
  g0[1] = lds_addr;                                    // LDS byte address
  g0[2] = (unsigned)(ga & 0xffffffffu);                // global_addr[31:0]
  g0[3] = (unsigned)((ga >> 32) & 0x1ffffffu)          // global_addr[56:32]
        | (2u << 30);                                  // type = 2 ("image")
  v8u g1;
  g1[0] = (1u << 16)                                   // data_size = 2 bytes
        | (1u << 20)                                   // pad_enable
        | (3u << 22)                                   // pad_interval: 16 DWORDs (64B)
        | (3u << 25);                                  // pad_amount: 4 DWORDs (16B)
  g1[1] = (tile_x & 0xffffu) << 16;                    // tensor_dim0[15:0]
  g1[2] = (tile_x >> 16) | ((tile_y & 0xffffu) << 16); // tensor_dim0[31:16], tensor_dim1[15:0]
  g1[3] = (tile_y >> 16) | (tile_x << 16);             // tensor_dim1[31:16], tile_dim0
  g1[4] = tile_y;                                      // tile_dim1, tile_dim2=0
  g1[5] = row_stride_elems;                            // tensor_dim0_stride[31:0]
  g1[6] = 0;                                           // stride[47:32], dim1_stride[15:0]
  g1[7] = 0;
  asm volatile("tensor_load_to_lds %0, %1" :: "s"(g0), "s"(g1) : "memory");
}

// ---------------- precompute kernels (tiny vs. the GEMMs) ----------------

// elementwise f32 -> bf16 (used for X and for W^T, which is proj_w as-is)
__global__ __launch_bounds__(256) void k_cvt_bf16(const float* __restrict__ x,
                                                  bf16_t* __restrict__ xb) {
  size_t i = (size_t)blockIdx.x * 256 + threadIdx.x;       // one float4 each
  const float4 v = ((const float4*)x)[i];
  bf16_t t[4] = {(bf16_t)v.x, (bf16_t)v.y, (bf16_t)v.z, (bf16_t)v.w};
  *(uint2*)(xb + 4 * i) = *(const uint2*)t;
}

__global__ __launch_bounds__(256) void k_norm_e(const float* __restrict__ sig,
                                                float* __restrict__ E) {
  __shared__ float red[256];
  const int k = blockIdx.x, tid = threadIdx.x;
  const float* s = sig + (size_t)k * DIMV;
  float acc = 0.f;
  for (int t = tid; t < DIMV; t += 256) { float v = s[t]; acc = fmaf(v, v, acc); }
  red[tid] = acc; __syncthreads();
  for (int o = 128; o > 0; o >>= 1) { if (tid < o) red[tid] += red[tid + o]; __syncthreads(); }
  const float inv = 1.0f / fmaxf(sqrtf(red[0]), 1e-12f);
  for (int t = tid; t < DIMV; t += 256) E[(size_t)k * DIMV + t] = s[t] * inv;
}

// L[k,t] = (1/d) * sum_f cos(p_k*theta[f] + 2*pi*f*t/d)    (direct IDFT, real part)
__global__ __launch_bounds__(256) void k_dft_l(const float* __restrict__ pos,
                                               const float* __restrict__ theta,
                                               float* __restrict__ L) {
  const int idx = blockIdx.x * 256 + threadIdx.x;          // NEXP*DIMV threads
  const int k = idx >> 12, t = idx & 4095;
  const float pk = pos[k];
  const float w0 = 6.28318530717958647692f * (float)t * (1.0f / 4096.0f);
  float s = 0.f;
  for (int f = 0; f < DIMV; ++f)
    s += __cosf(fmaf(pk, theta[f], w0 * (float)f));
  L[idx] = s * (1.0f / 4096.0f);
}

// R[t] = sum_k sum_tau E[k,tau] * L[k,(t-tau) mod d]
__global__ __launch_bounds__(256) void k_conv_r(const float* __restrict__ E,
                                                const float* __restrict__ L,
                                                float* __restrict__ R) {
  __shared__ float red[256];
  const int t = blockIdx.x, tid = threadIdx.x;
  float s = 0.f;
  for (int idx = tid; idx < NEXP * DIMV; idx += 256) {
    const int k = idx >> 12, tau = idx & 4095;
    s = fmaf(E[idx], L[((size_t)k << 12) + ((t - tau) & 4095)], s);
  }
  red[tid] = s; __syncthreads();
  for (int o = 128; o > 0; o >>= 1) { if (tid < o) red[tid] += red[tid + o]; __syncthreads(); }
  if (tid == 0) R[t] = red[0];
}

// C^T stored [t][tau]:  Ct[t*DIMV + tau] = bf16( R[(tau-t) mod d] )
__global__ __launch_bounds__(256) void k_build_ct(const float* __restrict__ R,
                                                  bf16_t* __restrict__ ct) {
  const size_t i4 = ((size_t)blockIdx.x * 256 + threadIdx.x) * 4;
  const int t = (int)(i4 >> 12), tau0 = (int)(i4 & 4095);
  bf16_t tmp[4];
#pragma unroll
  for (int e = 0; e < 4; ++e) tmp[e] = (bf16_t)R[(tau0 + e - t) & 4095];
  *(uint2*)(ct + i4) = *(const uint2*)tmp;
}

// CE[tau*NEXP + e] = sum_t R[(tau-t) mod d] * E[e,t]
__global__ __launch_bounds__(256) void k_ce(const float* __restrict__ R,
                                            const float* __restrict__ E,
                                            float* __restrict__ CE) {
  const int idx = blockIdx.x * 256 + threadIdx.x;          // DIMV*NEXP threads
  const int tau = idx >> 6, e = idx & 63;
  const float* Er = E + ((size_t)e << 12);
  float s = 0.f;
  for (int t = 0; t < DIMV; ++t) s = fmaf(R[(tau - t) & 4095], Er[t], s);
  CE[idx] = s;
}

// ---------------- WMMA GEMM: 128x128 block tile, K-step 32 ----------------
// A is [M][K] row-major; B is PRE-TRANSPOSED [N][K] row-major.
// A staged by TDM (wave 0, TENSORcnt), B by per-lane async DMA (ASYNCcnt);
// both double buffered.  LDS rows: 32 bf16 natural order + 16B pad (stride 40).
// MODE 0: write bf16 output Y.   MODE 1: per-row sum-of-squares partials.
template <int MODE>
__global__ __launch_bounds__(256)
void k_gemm_bf16(const bf16_t* __restrict__ A, const bf16_t* __restrict__ Bt,
                 bf16_t* __restrict__ Yout, float* __restrict__ partials,
                 int M, int N, int K) {
  __shared__ bf16_t lA[2][128 * 40];  // A tiles (natural k order per row)
  __shared__ bf16_t lB[2][128 * 40];  // B tiles, N-major rows [n][k0..31]
  __shared__ float  sq[128][33];      // deterministic row-sumsq reduction

  const int tid  = threadIdx.x;
  const int lane = tid & 31, w = tid >> 5;
  const int wm = w >> 1, wn = w & 1;          // 4x2 wave grid
  const int lrow = lane & 15, kh = lane >> 4; // frag row / K-half
  const int bm = (int)blockIdx.y * 128;
  const int bn = (int)blockIdx.x * 128;

  v8f acc[2][4];
#pragma unroll
  for (int i = 0; i < 2; ++i)
#pragma unroll
    for (int j = 0; j < 4; ++j) acc[i][j] = (v8f){};

  // B staging decomposition: 2 threads per tile row, two 16B chunks each
  const int arow = tid >> 1, ahalf = tid & 1;

  const unsigned lA_base = (unsigned)(uintptr_t)&lA[0][0];
  const unsigned lB_base = (unsigned)(uintptr_t)&lB[0][0];
  const unsigned bOff0 = (unsigned)(arow * 40 + ahalf * 16) * 2u;
  const unsigned bOff1 = bOff0 + 16u;
  const bf16_t* gA0 = A  + (size_t)bm * K;                    // TDM tile base
  const bf16_t* gB0 = Bt + (size_t)(bn + arow) * K + ahalf * 16;
  const unsigned BUFB = 128u * 40u * 2u;                       // bytes/buffer

  // one K-step of WMMAs on tile buffer `cur`
  auto compute = [&](int cur) {
    const bf16_t* curA = &lA[cur][0];
    const bf16_t* curB = &lB[cur][0];
    // A frag halves (natural row order): k {0..7,16..23} for kh=0,
    // k {8..15,24..31} for kh=1  -> bytes kh*16 and 32+kh*16 within row.
    v16bf af[2], bf[4];
#pragma unroll
    for (int i = 0; i < 2; ++i) {
      const bf16_t* pr = curA + (wm * 32 + i * 16 + lrow) * 40 + kh * 8;
      af[i] = load_frag16_2(pr, pr + 16);
    }
#pragma unroll
    for (int j = 0; j < 4; ++j) {
      const bf16_t* pr = curB + (wn * 64 + j * 16 + lrow) * 40 + kh * 8;
      bf[j] = load_frag16_2(pr, pr + 16);
    }
#pragma unroll
    for (int i = 0; i < 2; ++i)
#pragma unroll
      for (int j = 0; j < 4; ++j)
        acc[i][j] = __builtin_amdgcn_wmma_f32_16x16x32_bf16(
            false, af[i], false, bf[j], (short)0, acc[i][j], false, false);
  };

  // preload tile 0 into buffer 0
  if (w == 0) tdm_load_2d(lA_base, gA0, 32u, 128u, (unsigned)K);
  async_copy16(lB_base + bOff0, gB0, 0);
  async_copy16(lB_base + bOff1, gB0, 16);

  // steady state: branch-free body; tile i+1 streams during tile i's WMMAs
  int kb = 0;
  for (; kb < K - 32; kb += 32) {
    const int cur = (kb >> 5) & 1;
    const unsigned nb = (unsigned)(cur ^ 1) * BUFB;
    if (w == 0) tdm_load_2d(lA_base + nb, gA0 + kb + 32, 32u, 128u, (unsigned)K);
    const bf16_t* gb = gB0 + kb + 32;
    async_copy16(lB_base + nb + bOff0, gb, 0);
    async_copy16(lB_base + nb + bOff1, gb, 16);
    __builtin_prefetch(gB0 + kb + 64, 0, 0);
    // SALU waits run regardless of EXEC; waves 1-7 have TENSORcnt==0.
    // in-order completion: <=1 tensor / <=2 async outstanding => tile i landed
    __builtin_amdgcn_s_wait_tensorcnt(1);
    asm volatile("s_wait_asynccnt 0x2" ::: "memory");
    __syncthreads();   // cross-wave visibility of tile i
    compute(cur);
    __syncthreads();   // all reads of buffer `cur` done before it is refilled
  }
  // peeled last K-step: drain the pipeline
  {
    const int cur = (kb >> 5) & 1;
    __builtin_amdgcn_s_wait_tensorcnt(0);
    asm volatile("s_wait_asynccnt 0x0" ::: "memory");
    __syncthreads();
    compute(cur);
  }

  if (MODE == 0) {
    // D layout: VGPR v -> M = v + 8*kh, N = lrow (per 16x16 tile)
#pragma unroll
    for (int i = 0; i < 2; ++i)
#pragma unroll
      for (int j = 0; j < 4; ++j) {
        const int n = bn + wn * 64 + j * 16 + lrow;
#pragma unroll
        for (int v = 0; v < 8; ++v) {
          const int m = bm + wm * 32 + i * 16 + v + 8 * kh;
          Yout[(size_t)m * N + n] = (bf16_t)acc[i][j][v];
        }
      }
  } else {
    // deterministic per-row sum of squares of this block's S tile
#pragma unroll
    for (int i = 0; i < 2; ++i)
#pragma unroll
      for (int v = 0; v < 8; ++v) {
        float s = 0.f;
#pragma unroll
        for (int j = 0; j < 4; ++j) { float e = acc[i][j][v]; s = fmaf(e, e, s); }
        sq[wm * 32 + i * 16 + v + 8 * kh][wn * 16 + lrow] = s;
      }
    __syncthreads();
    if (tid < 128) {
      float s = 0.f;
#pragma unroll
      for (int c = 0; c < 32; ++c) s += sq[tid][c];
      partials[(size_t)(bm + tid) * gridDim.x + blockIdx.x] = s;
    }
  }
}

__global__ __launch_bounds__(256) void k_norm_reduce(const float* __restrict__ partials,
                                                     float* __restrict__ norm2) {
  const int r = blockIdx.x * 256 + threadIdx.x;
  float s = 0.f;
  for (int c = 0; c < 32; ++c) s += partials[(size_t)r * 32 + c];
  norm2[r] = s;
}

// P[tok*NEXP + e] = sum_t Y[tok,t] * CE[t,e]   (8.6 GFLOP, scalar is fine)
__global__ __launch_bounds__(256) void k_p(const bf16_t* __restrict__ Yb,
                                           const float* __restrict__ CE,
                                           float* __restrict__ P) {
  const int idx = blockIdx.x * 256 + threadIdx.x;
  const int tok = idx >> 6, e = idx & 63;
  const bf16_t* y = Yb + ((size_t)tok << 12);
  float s = 0.f;
  for (int t = 0; t < DIMV; ++t) s = fmaf((float)y[t], CE[t * NEXP + e], s);
  P[idx] = s;
}

// scores = P / max(||S||,1e-12); top-2; softmax(top2).  out = [w | idx | scores]
__global__ __launch_bounds__(256) void k_final(const float* __restrict__ P,
                                               const float* __restrict__ norm2,
                                               float* __restrict__ out) {
  const int tok = blockIdx.x * 256 + threadIdx.x;
  const float inv = 1.0f / fmaxf(sqrtf(fmaxf(norm2[tok], 0.f)), 1e-12f);
  float b0 = -3.0e38f, b1 = -3.0e38f; int i0 = 0, i1 = 0;
  float* sc = out + 4 * (size_t)N_TOK + (size_t)tok * NEXP;
  for (int e = 0; e < NEXP; ++e) {
    const float s = P[(size_t)tok * NEXP + e] * inv;
    sc[e] = s;
    if (s > b0) { b1 = b0; i1 = i0; b0 = s; i0 = e; }
    else if (s > b1) { b1 = s; i1 = e; }
  }
  const float e1 = __expf(b1 - b0), den = 1.0f + e1;
  out[(size_t)tok * 2 + 0] = 1.0f / den;
  out[(size_t)tok * 2 + 1] = e1 / den;
  out[2 * (size_t)N_TOK + tok * 2 + 0] = (float)i0;
  out[2 * (size_t)N_TOK + tok * 2 + 1] = (float)i1;
}

// ------------------------------- launcher --------------------------------

extern "C" void kernel_launch(void* const* d_in, const int* in_sizes, int n_in,
                              void* d_out, int out_size, void* d_ws, size_t ws_size,
                              hipStream_t stream) {
  const float* x   = (const float*)d_in[0];
  const float* pw  = (const float*)d_in[1];
  const float* pos = (const float*)d_in[2];
  const float* th  = (const float*)d_in[3];
  const float* sig = (const float*)d_in[4];
  float* out = (float*)d_out;

  char* ws = (char*)d_ws;
  size_t off = 0;
  auto take = [&](size_t bytes) -> void* {
    void* p = ws + off;
    off = (off + bytes + 255) & ~(size_t)255;
    return p;
  };
  bf16_t* Xb = (bf16_t*)take((size_t)N_TOK * HID * 2);    // 128 MB
  bf16_t* Wt = (bf16_t*)take((size_t)DIMV * HID * 2);     //  32 MB  (= bf16(proj_w), already [N][K])
  bf16_t* Ct = (bf16_t*)take((size_t)DIMV * DIMV * 2);    //  32 MB  (C^T, [N][K])
  bf16_t* Yb = (bf16_t*)take((size_t)N_TOK * DIMV * 2);   // 128 MB
  float*  E  = (float*)take((size_t)NEXP * DIMV * 4);
  float*  L  = (float*)take((size_t)NEXP * DIMV * 4);
  float*  R  = (float*)take((size_t)DIMV * 4);
  float*  CE = (float*)take((size_t)DIMV * NEXP * 4);
  float*  P  = (float*)take((size_t)N_TOK * NEXP * 4);
  float*  Pp = (float*)take((size_t)N_TOK * 32 * 4);      // norm partials
  float*  N2 = (float*)take((size_t)N_TOK * 4);

  // precompute
  k_cvt_bf16<<<(N_TOK * (size_t)HID) / 1024, 256, 0, stream>>>(x, Xb);
  k_cvt_bf16<<<((size_t)DIMV * HID) / 1024, 256, 0, stream>>>(pw, Wt);
  k_norm_e<<<NEXP, 256, 0, stream>>>(sig, E);
  k_dft_l<<<(NEXP * DIMV) / 256, 256, 0, stream>>>(pos, th, L);
  k_conv_r<<<DIMV, 256, 0, stream>>>(E, L, R);
  k_build_ct<<<((size_t)DIMV * DIMV) / 1024, 256, 0, stream>>>(R, Ct);
  k_ce<<<(DIMV * NEXP) / 256, 256, 0, stream>>>(R, E, CE);

  // GEMM1: Y = X @ W^T   (bf16 WMMA, f32 acc, TDM + async double buffering)
  k_gemm_bf16<0><<<dim3(DIMV / 128, N_TOK / 128), 256, 0, stream>>>(
      Xb, Wt, Yb, nullptr, N_TOK, DIMV, HID);
  // GEMM2: row ||Y @ C||^2 partials (S never materialized)
  k_gemm_bf16<1><<<dim3(DIMV / 128, N_TOK / 128), 256, 0, stream>>>(
      Yb, Ct, nullptr, Pp, N_TOK, DIMV, DIMV);
  k_norm_reduce<<<N_TOK / 256, 256, 0, stream>>>(Pp, N2);

  // score numerators + finalization
  k_p<<<(N_TOK * NEXP) / 256, 256, 0, stream>>>(Yb, CE, P);
  k_final<<<N_TOK / 256, 256, 0, stream>>>(P, N2, out);
}
